// SAGE_68805376082494
// MI455X (gfx1250) — compile-verified, matrix operand order checked by reading, not run on hardware
//
#include <hip/hip_runtime.h>
#include <hip/hip_bf16.h>

typedef __attribute__((ext_vector_type(16))) __bf16 v16bf;
typedef __attribute__((ext_vector_type(8)))  __bf16 v8bf;
typedef __attribute__((ext_vector_type(8)))  float  v8f;

static __device__ __forceinline__ unsigned short f32_to_bf16_rne(float f) {
    unsigned int u = __float_as_uint(f);
    u += 0x7FFFu + ((u >> 16) & 1u);           // round to nearest even
    return (unsigned short)(u >> 16);
}

// load one 16x32 bf16 WMMA fragment (A or B^T), row-major, stride Fin
static __device__ __forceinline__ v16bf load_frag(const unsigned short* base) {
    v8bf lo = *(const v8bf*)(base);            // K 0-7  / 8-15  (per lane half)
    v8bf hi = *(const v8bf*)(base + 16);       // K 16-23 / 24-31
    return __builtin_shufflevector(lo, hi, 0,1,2,3,4,5,6,7,8,9,10,11,12,13,14,15);
}

// ---------------------------------------------------------------- utilities
__global__ void k_zero_f32(float* __restrict__ p, int n4) {
    int i = blockIdx.x * blockDim.x + threadIdx.x;
    if (i < n4) ((float4*)p)[i] = make_float4(0.f, 0.f, 0.f, 0.f);
}

__global__ void k_zero_f32_scalar(float* __restrict__ p, int n) {
    int i = blockIdx.x * blockDim.x + threadIdx.x;
    if (i < n) p[i] = 0.f;
}

// ---------------------------------------------------------------- degree
__global__ void k_degree(const int* __restrict__ dst, int E, float* __restrict__ deg) {
    int i = blockIdx.x * blockDim.x + threadIdx.x;
    if (i < E) atomicAdd(&deg[dst[i]], 1.0f);
}

__global__ void k_inv_deg(float* __restrict__ deg, int n) {
    int i = blockIdx.x * blockDim.x + threadIdx.x;
    if (i < n) deg[i] = 1.0f / (deg[i] + 1.0f);
}

// --------------------------------------------------- edge scatter (segment_sum)
__global__ void k_scatter(const int* __restrict__ src, const int* __restrict__ dst,
                          const float* __restrict__ h, float* __restrict__ agg,
                          long long total, int gshift /* log2(F/4) */, int F) {
    long long t = (long long)blockIdx.x * blockDim.x + threadIdx.x;
    if (t >= total) return;
    int e = (int)(t >> gshift);
    int g = (int)(t & ((1 << gshift) - 1));
    int s = src[e], d = dst[e];
    float4 v = *(const float4*)(h + (long long)s * F + 4 * g);
    float* base = agg + (long long)d * F + 4 * g;
    atomicAdd(base + 0, v.x);
    atomicAdd(base + 1, v.y);
    atomicAdd(base + 2, v.z);
    atomicAdd(base + 3, v.w);
}

// ------------------------------ (agg + h) * inv_deg_p1, pack to bf16 A matrix
__global__ void k_norm_bf16(const float* __restrict__ agg, const float* __restrict__ h,
                            const float* __restrict__ inv, unsigned short* __restrict__ A,
                            int total, int fshift /* log2(F) */) {
    int i = blockIdx.x * blockDim.x + threadIdx.x;
    if (i >= total) return;
    int node = i >> fshift;
    float v = (agg[i] + h[i]) * inv[node];
    A[i] = f32_to_bf16_rne(v);
}

// -------------------------------- W [Fin x Fout] f32 -> Wt [Fout x Fin] bf16
__global__ void k_packW(const float* __restrict__ W, unsigned short* __restrict__ Wt,
                        int Fin, int Fout) {
    int i = blockIdx.x * blockDim.x + threadIdx.x;
    if (i >= Fin * Fout) return;
    int fin = i / Fout, fout = i % Fout;
    Wt[fout * Fin + fin] = f32_to_bf16_rne(W[i]);
}

// ---------------------------------------------------------------- WMMA GEMM
// out[Nn x Fout] = A[Nn x Fin](bf16) x Wt[Fout x Fin]^T (bf16) + bias, opt ReLU
// 4 waves/block; each wave computes a 32(M) x 32(N) register-blocked tile:
// 2 A frags x 2 B frags -> 4 WMMAs per K-step (halves loads per WMMA).
__global__ void k_gemm_bf16_wmma(const unsigned short* __restrict__ A,
                                 const unsigned short* __restrict__ Wt,
                                 const float* __restrict__ bias,
                                 float* __restrict__ out,
                                 int Nn, int Fin, int Fout, int do_relu) {
    const int lane    = threadIdx.x & 31;
    const int wave    = threadIdx.x >> 5;
    const int tile_m  = blockIdx.x * 4 + wave;     // 32-row tile index
    const int m0      = tile_m * 32;
    if (m0 >= Nn) return;                           // wave-uniform: EXEC stays full
    const int n0      = blockIdx.y * 32;
    const int halfsel = lane >> 4;                  // 0: lanes 0-15, 1: lanes 16-31
    const int l15     = lane & 15;
    const int koff    = halfsel * 8;                // per-ISA 16-bit A/B layout

    int row0 = m0 + l15;       if (row0 >= Nn) row0 = Nn - 1;   // clamp, no divergence
    int row1 = m0 + 16 + l15;  if (row1 >= Nn) row1 = Nn - 1;
    const int col0 = n0 + l15;
    const int col1 = n0 + 16 + l15;

    const unsigned short* a0p = A  + (long long)row0 * Fin + koff;
    const unsigned short* a1p = A  + (long long)row1 * Fin + koff;
    const unsigned short* b0p = Wt + (long long)col0 * Fin + koff;
    const unsigned short* b1p = Wt + (long long)col1 * Fin + koff;

    v8f acc00 = {}, acc01 = {}, acc10 = {}, acc11 = {};
    for (int k = 0; k < Fin; k += 32) {
        v16bf a0 = load_frag(a0p + k);
        v16bf a1 = load_frag(a1p + k);
        v16bf b0 = load_frag(b0p + k);
        v16bf b1 = load_frag(b1p + k);
        acc00 = __builtin_amdgcn_wmma_f32_16x16x32_bf16(false, a0, false, b0,
                                                        (short)0, acc00, false, false);
        acc01 = __builtin_amdgcn_wmma_f32_16x16x32_bf16(false, a0, false, b1,
                                                        (short)0, acc01, false, false);
        acc10 = __builtin_amdgcn_wmma_f32_16x16x32_bf16(false, a1, false, b0,
                                                        (short)0, acc10, false, false);
        acc11 = __builtin_amdgcn_wmma_f32_16x16x32_bf16(false, a1, false, b1,
                                                        (short)0, acc11, false, false);
    }

    const float bn0 = bias[col0];
    const float bn1 = bias[col1];
#pragma unroll
    for (int r = 0; r < 8; ++r) {
        // C/D layout: lane half selects M+8; VGPR index r selects M%8
        int mA = m0 + r + halfsel * 8;          // rows of acc0x (M 0-15 of tile)
        int mB = mA + 16;                       // rows of acc1x (M 16-31 of tile)
        float v00 = acc00[r] + bn0;
        float v01 = acc01[r] + bn1;
        float v10 = acc10[r] + bn0;
        float v11 = acc11[r] + bn1;
        if (do_relu) {
            v00 = v00 > 0.f ? v00 : 0.f;  v01 = v01 > 0.f ? v01 : 0.f;
            v10 = v10 > 0.f ? v10 : 0.f;  v11 = v11 > 0.f ? v11 : 0.f;
        }
        if (mA < Nn) {
            out[(long long)mA * Fout + col0] = v00;
            out[(long long)mA * Fout + col1] = v01;
        }
        if (mB < Nn) {
            out[(long long)mB * Fout + col0] = v10;
            out[(long long)mB * Fout + col1] = v11;
        }
    }
}

// ------------------------------------------------- log_softmax over C=64/row
__global__ void k_log_softmax64(float* __restrict__ x, int Nn) {
    int row  = blockIdx.x * (blockDim.x >> 5) + (threadIdx.x >> 5);
    int lane = threadIdx.x & 31;
    if (row >= Nn) return;
    float* p = x + (long long)row * 64;
    float a0 = p[lane], a1 = p[lane + 32];
    float m = fmaxf(a0, a1);
    for (int o = 16; o > 0; o >>= 1) m = fmaxf(m, __shfl_xor(m, o, 32));
    float s = __expf(a0 - m) + __expf(a1 - m);
    for (int o = 16; o > 0; o >>= 1) s += __shfl_xor(s, o, 32);
    float lz = m + __logf(s);
    p[lane]      = a0 - lz;
    p[lane + 32] = a1 - lz;
}

// ===========================================================================
static inline size_t alignUp(size_t x) { return (x + 255) & ~(size_t)255; }

extern "C" void kernel_launch(void* const* d_in, const int* in_sizes, int n_in,
                              void* d_out, int out_size, void* d_ws, size_t ws_size,
                              hipStream_t stream) {
    const float* feat = (const float*)d_in[0];
    const int*   src  = (const int*)  d_in[1];
    const int*   dst  = (const int*)  d_in[2];
    const float* W0   = (const float*)d_in[3];
    const float* b0   = (const float*)d_in[4];
    const float* W1   = (const float*)d_in[5];
    const float* b1   = (const float*)d_in[6];
    const float* W2   = (const float*)d_in[7];
    const float* b2   = (const float*)d_in[8];

    const int H    = in_sizes[4];             // 256
    const int F_IN = in_sizes[3] / H;         // 128
    const int C    = in_sizes[8];             // 64
    const int Nn   = in_sizes[0] / F_IN;      // 50000
    const int E    = in_sizes[1];             // 600000

    // ---------------- workspace carve-up
    char* ws = (char*)d_ws;
    float* inv = (float*)ws;                ws += alignUp((size_t)Nn * 4);
    float* agg = (float*)ws;                ws += alignUp((size_t)Nn * H * 4);
    float* h1  = (float*)ws;                ws += alignUp((size_t)Nn * H * 4);
    unsigned short* Abf = (unsigned short*)ws; ws += alignUp((size_t)Nn * H * 2);
    unsigned short* Wt0 = (unsigned short*)ws; ws += alignUp((size_t)F_IN * H * 2);
    unsigned short* Wt1 = (unsigned short*)ws; ws += alignUp((size_t)H * H * 2);
    unsigned short* Wt2 = (unsigned short*)ws; ws += alignUp((size_t)H * C * 2);

    const int T = 256;
    float* out = (float*)d_out;

    // ---------------- degree -> inv(deg+1)
    k_zero_f32_scalar<<<(Nn + T - 1) / T, T, 0, stream>>>(inv, Nn);
    k_degree<<<(E + T - 1) / T, T, 0, stream>>>(dst, E, inv);
    k_inv_deg<<<(Nn + T - 1) / T, T, 0, stream>>>(inv, Nn);

    // ---------------- pack weights (transposed bf16)
    k_packW<<<(F_IN * H + T - 1) / T, T, 0, stream>>>(W0, Wt0, F_IN, H);
    k_packW<<<(H * H + T - 1) / T, T, 0, stream>>>(W1, Wt1, H, H);
    k_packW<<<(H * C + T - 1) / T, T, 0, stream>>>(W2, Wt2, H, C);

    const int tilesM = (Nn + 31) / 32;        // 32-row tiles
    const int gx     = (tilesM + 3) / 4;      // 4 waves per block

    // ================ layer 0: F_IN -> H, relu
    {
        int tot = Nn * F_IN;
        k_zero_f32<<<(tot / 4 + T - 1) / T, T, 0, stream>>>(agg, tot / 4);
        long long stot = (long long)E * (F_IN / 4);
        k_scatter<<<(int)((stot + T - 1) / T), T, 0, stream>>>(src, dst, feat, agg,
                                                               stot, 5 /*log2(32)*/, F_IN);
        k_norm_bf16<<<(tot + T - 1) / T, T, 0, stream>>>(agg, feat, inv, Abf, tot, 7);
        dim3 g(gx, H / 32);
        k_gemm_bf16_wmma<<<g, 128, 0, stream>>>(Abf, Wt0, b0, h1, Nn, F_IN, H, 1);
    }

    // ================ layer 1: H -> H, relu
    {
        int tot = Nn * H;
        k_zero_f32<<<(tot / 4 + T - 1) / T, T, 0, stream>>>(agg, tot / 4);
        long long stot = (long long)E * (H / 4);
        k_scatter<<<(int)((stot + T - 1) / T), T, 0, stream>>>(src, dst, h1, agg,
                                                               stot, 6 /*log2(64)*/, H);
        k_norm_bf16<<<(tot + T - 1) / T, T, 0, stream>>>(agg, h1, inv, Abf, tot, 8);
        dim3 g(gx, H / 32);
        k_gemm_bf16_wmma<<<g, 128, 0, stream>>>(Abf, Wt1, b1, h1, Nn, H, H, 1);
    }

    // ================ layer 2: H -> C, then log_softmax
    {
        int tot = Nn * H;
        k_zero_f32<<<(tot / 4 + T - 1) / T, T, 0, stream>>>(agg, tot / 4);
        long long stot = (long long)E * (H / 4);
        k_scatter<<<(int)((stot + T - 1) / T), T, 0, stream>>>(src, dst, h1, agg,
                                                               stot, 6, H);
        k_norm_bf16<<<(tot + T - 1) / T, T, 0, stream>>>(agg, h1, inv, Abf, tot, 8);
        dim3 g(gx, C / 32);
        k_gemm_bf16_wmma<<<g, 128, 0, stream>>>(Abf, Wt2, b2, out, Nn, H, C, 0);
    }

    const int rowsPerBlk = T / 32;
    k_log_softmax64<<<(Nn + rowsPerBlk - 1) / rowsPerBlk, T, 0, stream>>>(out, Nn);
}
